// GGMLLayer_283467842170
// MI455X (gfx1250) — matrix operand order repaired and checked by reading, not compiled, not run on hardware
//
#include <hip/hip_runtime.h>

typedef __attribute__((ext_vector_type(16))) _Float16 v16h;
typedef __attribute__((ext_vector_type(8)))  _Float16 v8h;
typedef __attribute__((ext_vector_type(8)))  float    v8f;

// ---------------------------------------------------------------------------
// Pre-pass: convert activations x [32, I] f32 -> f16 in workspace.
// x is tiny (512 KB) and reused by all 688 N-tiles, so converting once saves
// per-use cvt work and halves A-fetch bytes (and A loads hit L2 every time).
// ---------------------------------------------------------------------------
__global__ void cvt_x_f16(const float* __restrict__ x,
                          _Float16* __restrict__ xh, int n) {
    int i = blockIdx.x * blockDim.x + threadIdx.x;
    if (i < n) xh[i] = (_Float16)x[i];
}

// ---------------------------------------------------------------------------
// Fused dequant + GEMM:  y[m, o] = sum_k x[m,k] * (q[o,k]-127)*scale[o,k/32] + b[o]
//
// One workgroup per 16-column output tile. 8 waves split K (4096 -> 512 each,
// = 16 Q8_0 blocks). WMMA f16 16x16x32: K exactly one quant block, so the
// per-block scale is applied in f32 on the WMMA result (per-lane scalar,
// since C-layout lane == output column). q-127 is exact in f16.
// ---------------------------------------------------------------------------
__global__ __launch_bounds__(256) void qlinear_wmma(
    const _Float16* __restrict__ xh,     // [32, I] f16
    const int*      __restrict__ qw,     // [O, I] int32 quants 0..255
    const float*    __restrict__ scales, // [O, I/32]
    const float*    __restrict__ bias,   // [O]
    float*          __restrict__ y,      // [32, O]
    int I, int O, int nb, int kbPerWave)
{
    __shared__ float red[8][2][8][32];   // 16 KB: [wave][mtile][c-vgpr][lane]

    const int tid  = threadIdx.x;
    const int wave = tid >> 5;
    const int lane = tid & 31;
    const int n    = lane & 15;          // column within tile / A row within tile
    const int hi   = lane >> 4;          // lane half-group
    const int obase = blockIdx.x * 16;
    const int o     = obase + n;         // this lane's output column

    // ISA VGPR layouts (CDNA5 ISA):
    //  A 16x32 f16: lanes 0-15 hold K 0-7 & 16-23, lanes 16-31 hold K 8-15 & 24-31
    //  B 32x16 f16: lanes 0-15 hold K 0-7 & 16-23 (col=lane), lanes 16-31 hold K 8-15 & 24-31
    const int klo_a = hi ? 8  : 0;
    const int klo_b = hi ? 8  : 0;

    const int kb0 = wave * kbPerWave;

    v8f acc0 = {};  // M rows 0..15
    v8f acc1 = {};  // M rows 16..31

    const int*      wrow  = qw     + (size_t)o * I;
    const float*    srow  = scales + (size_t)o * nb;
    const _Float16* arow0 = xh + (size_t)n        * I;
    const _Float16* arow1 = xh + (size_t)(n + 16) * I;

    for (int kk = 0; kk < kbPerWave; ++kk) {
        const int kb = kb0 + kk;
        const int k  = kb * 32;

        // ---- A tiles: two 16-B loads per tile (K klo..klo+7, klo+16..klo+23)
        union { v16h v; v8h h[2]; } a0, a1;
        const v8h* pa0 = (const v8h*)(arow0 + k + klo_a);
        const v8h* pa1 = (const v8h*)(arow1 + k + klo_a);
        a0.h[0] = pa0[0]; a0.h[1] = pa0[2];
        a1.h[0] = pa1[0]; a1.h[1] = pa1[2];

        // ---- B tile: 16 int32 quants (K klo..klo+7, klo+16..klo+23), 4 x B128, 64 B/lane
        const int4* pb = (const int4*)(wrow + k + klo_b);
        int4 q0 = pb[0], q1 = pb[1], q2 = pb[4], q3 = pb[5];

        v16h b;
        b[ 0] = (_Float16)(q0.x - 127); b[ 1] = (_Float16)(q0.y - 127);
        b[ 2] = (_Float16)(q0.z - 127); b[ 3] = (_Float16)(q0.w - 127);
        b[ 4] = (_Float16)(q1.x - 127); b[ 5] = (_Float16)(q1.y - 127);
        b[ 6] = (_Float16)(q1.z - 127); b[ 7] = (_Float16)(q1.w - 127);
        b[ 8] = (_Float16)(q2.x - 127); b[ 9] = (_Float16)(q2.y - 127);
        b[10] = (_Float16)(q2.z - 127); b[11] = (_Float16)(q2.w - 127);
        b[12] = (_Float16)(q3.x - 127); b[13] = (_Float16)(q3.y - 127);
        b[14] = (_Float16)(q3.z - 127); b[15] = (_Float16)(q3.w - 127);

        v8f z = {};
        v8f c0 = __builtin_amdgcn_wmma_f32_16x16x32_f16(
                     false, a0.v, false, b, (short)0, z, false, false);
        v8f c1 = __builtin_amdgcn_wmma_f32_16x16x32_f16(
                     false, a1.v, false, b, (short)0, z, false, false);

        // per-block, per-column scale: per-lane scalar in C layout
        float s = srow[kb];
        acc0 += c0 * s;
        acc1 += c1 * s;
    }

    // ---- cross-wave K reduction via LDS
    #pragma unroll
    for (int i = 0; i < 8; ++i) {
        red[wave][0][i][lane] = acc0[i];
        red[wave][1][i][lane] = acc1[i];
    }
    __syncthreads();

    // 512 outputs (32 rows x 16 cols), 256 threads -> 2 each.
    // C layout: VGPR i, lanes 0-15: M=i, N=lane; lanes 16-31: M=8+i, N=lane-16.
    for (int slot = tid; slot < 512; slot += 256) {
        int t = slot >> 8;
        int i = (slot >> 5) & 7;
        int l = slot & 31;
        float sum = 0.f;
        #pragma unroll
        for (int w = 0; w < 8; ++w) sum += red[w][t][i][l];
        int m  = t * 16 + ((l < 16) ? i : (i + 8));
        int oo = obase + (l & 15);
        y[(size_t)m * O + oo] = sum + bias[oo];
    }
}

extern "C" void kernel_launch(void* const* d_in, const int* in_sizes, int n_in,
                              void* d_out, int out_size, void* d_ws, size_t ws_size,
                              hipStream_t stream) {
    const float* x      = (const float*)d_in[0];
    const int*   qw     = (const int*)d_in[1];
    const float* scales = (const float*)d_in[2];
    const float* bias   = (const float*)d_in[3];
    float* y = (float*)d_out;
    _Float16* xh = (_Float16*)d_ws;

    const int M  = 32;                 // batch rows (fixed by reference)
    const int nx = in_sizes[0];        // 32 * I
    const int I  = nx / M;             // 4096
    const int O  = in_sizes[3];        // 11008
    const int nb = I / 32;             // quant blocks per row = 128
    const int kbPerWave = nb / 8;      // 16 blocks per wave

    cvt_x_f16<<<(nx + 255) / 256, 256, 0, stream>>>(x, xh, nx);
    qlinear_wmma<<<O / 16, 256, 0, stream>>>(xh, qw, scales, bias, y,
                                             I, O, nb, kbPerWave);
}
